// Architecture_1365799600741
// MI455X (gfx1250) — compile-verified, matrix operand order checked by reading, not run on hardware
//
#include <hip/hip_runtime.h>
#include <hip/hip_bf16.h>

typedef __attribute__((ext_vector_type(16))) __bf16 v16bf;
typedef __attribute__((ext_vector_type(8)))  __bf16 v8bf;
typedef __attribute__((ext_vector_type(8)))  float  v8f;

#define Bc 32
#define Hc 8
#define Sc 512
#define Dc 64
#define QT 32              // query rows per workgroup
#define SSTRIDE 516        // f32 elems per score row (pad: lane halves hit disjoint banks)
#define PSTRIDE 1032       // bf16 elems per P row (aliases one score row exactly: 516*4 bytes)
#define VSTRIDE 520        // bf16 elems per Vt row
#define NEGF (-1e32f)
#define FLTMAX 3.402823466e38f

__device__ __forceinline__ float wave_max(float v) {
    #pragma unroll
    for (int o = 16; o; o >>= 1) v = fmaxf(v, __shfl_xor(v, o, 32));
    return v;
}
__device__ __forceinline__ float wave_sum(float v) {
    #pragma unroll
    for (int o = 16; o; o >>= 1) v += __shfl_xor(v, o, 32);
    return v;
}

// Build a 16x32 bf16 A/B fragment row-slice from a row-major f32 row.
// Lane holds K = kb+0..7 and kb+16..23 (kb = 0 or 8 by lane half).
__device__ __forceinline__ v16bf make_frag(const float* __restrict__ rowp, int kb) {
    const float4* p = (const float4*)rowp;
    float4 f0 = p[(kb >> 2) + 0];
    float4 f1 = p[(kb >> 2) + 1];
    float4 g0 = p[(kb >> 2) + 4];
    float4 g1 = p[(kb >> 2) + 5];
    v16bf a;
    a[0]=(__bf16)f0.x; a[1]=(__bf16)f0.y; a[2]=(__bf16)f0.z; a[3]=(__bf16)f0.w;
    a[4]=(__bf16)f1.x; a[5]=(__bf16)f1.y; a[6]=(__bf16)f1.z; a[7]=(__bf16)f1.w;
    a[8]=(__bf16)g0.x; a[9]=(__bf16)g0.y; a[10]=(__bf16)g0.z; a[11]=(__bf16)g0.w;
    a[12]=(__bf16)g1.x; a[13]=(__bf16)g1.y; a[14]=(__bf16)g1.z; a[15]=(__bf16)g1.w;
    return a;
}

__device__ __forceinline__ v16bf lds_frag(const __bf16* __restrict__ rowp) {
    const v8bf* p = (const v8bf*)rowp;
    v8bf lo = p[0];   // elems 0..7
    v8bf hi = p[2];   // elems 16..23
    v16bf a;
    #pragma unroll
    for (int j = 0; j < 8; j++) { a[j] = lo[j]; a[8 + j] = hi[j]; }
    return a;
}

__global__ __launch_bounds__(256, 1)
void topk_attn_gfx1250(const float* __restrict__ Q, const float* __restrict__ K,
                       const float* __restrict__ V, const int* __restrict__ maskp,
                       const int* __restrict__ dkp, const int* __restrict__ kidxp,
                       float* __restrict__ Out)
{
    (void)maskp; // mask is causal tril; applied from indices
    extern __shared__ char smem[];
    float*  Ssc = (float*)smem;                          // [QT][SSTRIDE] f32 scores
    __bf16* Pp  = (__bf16*)smem;                         // [QT][PSTRIDE] bf16 probs (aliases Ssc row-for-row)
    __bf16* Vt  = (__bf16*)(smem + QT * SSTRIDE * 4);    // [Dc][VSTRIDE] bf16 V^T

    const int bh   = blockIdx.x >> 4;           // Sc/QT == 16 q-tiles per head
    const int q0   = (blockIdx.x & 15) * QT;
    const int tid  = threadIdx.x;
    const int lane = tid & 31;
    const int w    = tid >> 5;                  // wave id 0..7
    const int rl   = lane & 15;                 // row/col within 16-tile
    const int hb   = lane >> 4;                 // lane half
    const int kb   = hb * 8;                    // K sub-offset for fragments

    const int   kidx  = kidxp[0];
    const float scale = rsqrtf((float)dkp[0]);
    const size_t base = (size_t)bh * Sc * Dc;

    // ---- Stage V^T into LDS as bf16 (coalesced global reads) ----
    for (int t = tid; t < Sc * Dc; t += 256) {
        int s = t >> 6, d = t & 63;
        Vt[d * VSTRIDE + s] = (__bf16)V[base + t];
    }

    // ---- Phase A: scores = scale * Q K^T with causal mask -> LDS f32 ----
    // Each wave owns 4 unique key tiles (nt = w + 8i) and computes BOTH
    // q-subtiles against them, so every K fragment feeds 2 WMMAs.
    const float* Qrow0 = Q + base + (size_t)(q0 + rl) * Dc;        // qt=0 rows
    const float* Qrow1 = Q + base + (size_t)(q0 + 16 + rl) * Dc;   // qt=1 rows
    v16bf aq00 = make_frag(Qrow0,      kb);   // qt0, d 0..31
    v16bf aq01 = make_frag(Qrow0 + 32, kb);   // qt0, d 32..63
    v16bf aq10 = make_frag(Qrow1,      kb);   // qt1, d 0..31
    v16bf aq11 = make_frag(Qrow1 + 32, kb);   // qt1, d 32..63
    #pragma unroll
    for (int i = 0; i < 4; i++) {
        int nt = w + 8 * i;                 // unique key tile per wave
        const float* Krow = K + base + (size_t)(nt * 16 + rl) * Dc;
        v16bf b0 = make_frag(Krow,      kb);
        v16bf b1 = make_frag(Krow + 32, kb);
        v8f c0 = {}, c1 = {};
        c0 = __builtin_amdgcn_wmma_f32_16x16x32_bf16(false, aq00, false, b0, (short)0, c0, false, false);
        c0 = __builtin_amdgcn_wmma_f32_16x16x32_bf16(false, aq01, false, b1, (short)0, c0, false, false);
        c1 = __builtin_amdgcn_wmma_f32_16x16x32_bf16(false, aq10, false, b0, (short)0, c1, false, false);
        c1 = __builtin_amdgcn_wmma_f32_16x16x32_bf16(false, aq11, false, b1, (short)0, c1, false, false);
        int kg = nt * 16 + rl;
        #pragma unroll
        for (int r = 0; r < 8; r++) {
            int ql0 = hb * 8 + r;           // qt=0 tile rows
            float v0 = c0[r] * scale;
            if (kg > q0 + ql0) v0 = NEGF;
            Ssc[ql0 * SSTRIDE + kg] = v0;
            int ql1 = 16 + hb * 8 + r;      // qt=1 tile rows
            float v1 = c1[r] * scale;
            if (kg > q0 + ql1) v1 = NEGF;
            Ssc[ql1 * SSTRIDE + kg] = v1;
        }
    }
    __syncthreads();

    // ---- Phase B/C/D fused: softmax1 -> top-k threshold -> softmax2 -> P bf16 ----
    for (int j = 0; j < 4; j++) {
        int row = w * 4 + j;
        int qg  = q0 + row;
        float x[16];
        #pragma unroll
        for (int i = 0; i < 16; i++) x[i] = Ssc[row * SSTRIDE + lane + 32 * i];

        float m1 = -FLTMAX;
        #pragma unroll
        for (int i = 0; i < 16; i++) m1 = fmaxf(m1, x[i]);
        m1 = wave_max(m1);
        float s1 = 0.f;
        #pragma unroll
        for (int i = 0; i < 16; i++) { x[i] = __expf(x[i] - m1); s1 += x[i]; }
        s1 = wave_sum(s1);
        float inv1 = 1.0f / s1;
        #pragma unroll
        for (int i = 0; i < 16; i++) x[i] *= inv1;       // softmax1 probs

        float m2 = 0.f;
        #pragma unroll
        for (int i = 0; i < 16; i++) m2 = fmaxf(m2, x[i]);
        m2 = wave_max(m2);

        float thr = -FLTMAX;
        if (qg >= kidx) {
            float c2[16];
            #pragma unroll
            for (int i = 0; i < 16; i++) c2[i] = x[i];
            for (int t = 0; t < kidx; t++) {
                float bv = -FLTMAX; int bi = 0;
                #pragma unroll
                for (int i = 0; i < 16; i++) if (c2[i] > bv) { bv = c2[i]; bi = i; }
                float v = bv; int wl = lane;
                #pragma unroll
                for (int o = 16; o; o >>= 1) {
                    float v2 = __shfl_xor(v, o, 32);
                    int   l2 = __shfl_xor(wl, o, 32);
                    if (v2 > v || (v2 == v && l2 < wl)) { v = v2; wl = l2; }
                }
                thr = v;                          // t-th largest so far
                if (lane == wl) c2[bi] = -FLTMAX; // deactivate exactly one
            }
        }

        float e2[16];
        float s2 = 0.f;
        #pragma unroll
        for (int i = 0; i < 16; i++) {
            bool keep = (qg < kidx) || (x[i] >= thr);
            e2[i] = keep ? __expf(x[i] - m2) : 0.f;
            s2 += e2[i];
        }
        s2 = wave_sum(s2);
        float inv2 = (qg == 0) ? 0.f : (1.0f / s2);  // pad_zero row 0
        #pragma unroll
        for (int i = 0; i < 16; i++)
            Pp[row * PSTRIDE + lane + 32 * i] = (__bf16)(e2[i] * inv2);
    }
    __syncthreads();

    // ---- Phase E: O = P V  (one 16x16 tile per wave, K=512) ----
    {
        const int qt2 = w & 1;
        const int dt  = w >> 1;                 // D tile 0..3
        v8f c = {};
        const __bf16* Pr = Pp + (size_t)(qt2 * 16 + rl) * PSTRIDE + kb;
        const __bf16* Vr = Vt + (size_t)(dt * 16 + rl) * VSTRIDE + kb;
        #pragma unroll
        for (int t = 0; t < 16; t++) {
            v16bf a = lds_frag(Pr + t * 32);
            v16bf b = lds_frag(Vr + t * 32);
            c = __builtin_amdgcn_wmma_f32_16x16x32_bf16(false, a, false, b, (short)0, c, false, false);
        }
        #pragma unroll
        for (int r = 0; r < 8; r++) {
            int qg = q0 + qt2 * 16 + hb * 8 + r;
            int d  = dt * 16 + rl;
            Out[base + (size_t)qg * Dc + d] = c[r];
        }
    }
}

extern "C" void kernel_launch(void* const* d_in, const int* in_sizes, int n_in,
                              void* d_out, int out_size, void* d_ws, size_t ws_size,
                              hipStream_t stream) {
    (void)in_sizes; (void)n_in; (void)out_size; (void)d_ws; (void)ws_size;
    const float* q    = (const float*)d_in[0];
    const float* k    = (const float*)d_in[1];
    const float* v    = (const float*)d_in[2];
    const int*   mask = (const int*)d_in[3];
    const int*   dk   = (const int*)d_in[4];
    const int*   kidx = (const int*)d_in[5];
    float* out = (float*)d_out;

    const size_t shmem = (size_t)QT * SSTRIDE * 4 + (size_t)Dc * VSTRIDE * 2; // 132,608 B
    (void)hipFuncSetAttribute((const void*)topk_attn_gfx1250,
                              hipFuncAttributeMaxDynamicSharedMemorySize, (int)shmem);
    dim3 grid(Bc * Hc * (Sc / QT));   // 4096 workgroups
    topk_attn_gfx1250<<<grid, 256, shmem, stream>>>(q, k, v, mask, dk, kidx, out);
}
